// LSTMModel_12403865550936
// MI455X (gfx1250) — matrix operand order repaired and checked by reading, not compiled, not run on hardware
//
#include <hip/hip_runtime.h>
#include <hip/hip_bf16.h>
#include <stdint.h>
#include <stddef.h>

// ---------------------------------------------------------------------------
// 2-layer LSTM (B=256, T=512, I=64, H=256) + FC(H->1) for gfx1250 (MI455X)
//   * bf16 WMMA (v_wmma_f32_16x16x32_bf16) everywhere, fp32 accumulate
//   * input projections as large parallel WMMA GEMMs, output stored in
//     WMMA C-fragment order so the scan seeds accumulators with one 32B load
//   * recurrence: persistent 32-WG kernel, Whh held in registers across all
//     512 steps, grid barrier per step, c-state resident in LDS
// ---------------------------------------------------------------------------

typedef __attribute__((ext_vector_type(16))) __bf16 v16bf;
typedef __attribute__((ext_vector_type(8)))  __bf16 v8bf;
typedef __attribute__((ext_vector_type(8)))  float  v8f;

#define B_  256
#define T_  512
#define I_  64
#define H_  256
#define G_  1024   // 4*H

__device__ __forceinline__ unsigned short f32_bf16(float f) {
  unsigned int u = __float_as_uint(f);
  unsigned int r = u + 0x7fffu + ((u >> 16) & 1u);   // round-to-nearest-even
  return (unsigned short)(r >> 16);
}

// A-matrix 16x32 bf16 fragment: lane l -> row l%16,
// K = 16*(l/16)+e contiguous in memory (matches B fragment).
__device__ __forceinline__ v16bf load_a_frag(const unsigned short* base, int row,
                                             int ldk, int kk, int lane) {
  const int half = lane >> 4;
  const unsigned short* p = base + (size_t)row * ldk + kk + 16 * half;
  union { v16bf v; v8bf h[2]; } u;
  u.h[0] = *reinterpret_cast<const v8bf*>(p);
  u.h[1] = *reinterpret_cast<const v8bf*>(p + 8);
  return u.v;
}

// B-matrix 32x16 bf16 fragment: lane l -> col l%16, K = 16*(l/16)+e contiguous.
// W stored row-major (gateCol, K) => one aligned 32B load per fragment.
__device__ __forceinline__ v16bf load_b_frag(const unsigned short* w, int ncol0,
                                             int ldk, int kk, int lane) {
  const int n = lane & 15, kh = lane >> 4;
  const unsigned short* p = w + (size_t)(ncol0 + n) * ldk + kk + 16 * kh;
  return *reinterpret_cast<const v16bf*>(p);
}

// ------------------------------- conversions -------------------------------
__global__ void cvt_f32_bf16(const float* __restrict__ in,
                             unsigned short* __restrict__ out, int n) {
  int i = blockIdx.x * blockDim.x + threadIdx.x;
  int stride = gridDim.x * blockDim.x;
  for (; i < n; i += stride) out[i] = f32_bf16(in[i]);
}

// x (B,T,I) f32  ->  x_bf (T,B,I) bf16   (t-major rows: m = t*B + b)
__global__ void cvt_permute_x(const float* __restrict__ in,
                              unsigned short* __restrict__ out) {
  int i = blockIdx.x * blockDim.x + threadIdx.x;
  int stride = gridDim.x * blockDim.x;
  const int n = B_ * T_ * I_;
  for (; i < n; i += stride) {
    const int d = i % I_;
    const int t = (i / I_) % T_;
    const int b = i / (I_ * T_);
    out[((size_t)t * B_ + b) * I_ + d] = f32_bf16(in[i]);
  }
}

// ------------------------- input-projection GEMM ---------------------------
// A rows are m = t*B + b (ldk = Kdim).  Output is stored in WMMA C-fragment
// order:  out[(((t*1024) + ct*16 + rr)*32 + lane)*8 + r]
// where ct = column tile (n/16, 0..63), rr = row tile within step (b/16).
__global__ __launch_bounds__(256) void xproj_gemm(
    const unsigned short* __restrict__ A, const unsigned short* __restrict__ W,
    const float* __restrict__ bih, const float* __restrict__ bhh,
    float* __restrict__ out, int Kdim) {
  const int lane = threadIdx.x & 31;
  const int wv   = threadIdx.x >> 5;
  const int gw   = blockIdx.x * 8 + wv;      // 131072 waves total
  const int rowTile = gw >> 4;               // 8192 m-tiles: t = rowTile/16
  const int colGrp  = gw & 15;               // 16 col groups of 64
  const int n0   = colGrp * 64;
  const int t    = rowTile >> 4;
  const int rr   = rowTile & 15;
  const int arow = rowTile * 16 + (lane & 15);

  v8f acc[4];
  for (int c = 0; c < 4; ++c)
    for (int r = 0; r < 8; ++r) acc[c][r] = 0.f;

  for (int kk = 0; kk < Kdim; kk += 32) {
    v16bf a = load_a_frag(A, arow, Kdim, kk, lane);
    for (int c = 0; c < 4; ++c) {
      v16bf b = load_b_frag(W, n0 + c * 16, Kdim, kk, lane);
      acc[c] = __builtin_amdgcn_wmma_f32_16x16x32_bf16(
          false, a, false, b, (short)0, acc[c], false, false);
    }
  }

  for (int c = 0; c < 4; ++c) {
    const int ct  = colGrp * 4 + c;
    const int col = ct * 16 + (lane & 15);
    const float bv = bih[col] + bhh[col];
    for (int r = 0; r < 8; ++r) acc[c][r] += bv;
    float* dst = out + (((size_t)t * 1024 + ct * 16 + rr) * 32 + lane) * 8;
    *reinterpret_cast<v8f*>(dst) = acc[c];   // contiguous 32B store
  }
}

// ----------------------------- recurrent scan ------------------------------
// Persistent kernel: 32 WGs x 1024 threads (32 waves).  Whh fragments for this
// wave's 16 gate-columns are preloaded into 64 VGPRs and reused for all 512
// steps.  Each step: gates = h @ Whh^T + xproj[t]; LSTM cell; grid barrier.
__global__ __launch_bounds__(1024) void lstm_scan(
    const float* __restrict__ xproj, const unsigned short* __restrict__ Whh,
    unsigned short* __restrict__ hbuf,      // 2 * B_*H_ bf16 (ping-pong)
    unsigned short* __restrict__ seqOut,    // (T,B,H) bf16 row=(t*B+b), or null
    float* __restrict__ lastH,              // (B,H) fp32, or null
    unsigned int* __restrict__ bar) {
  __shared__ float lds_g[4][128][16];       // 32 KB gate staging
  __shared__ float lds_c[128][16];          // 8 KB persistent c-state

  const int lane    = threadIdx.x & 31;
  const int wv      = threadIdx.x >> 5;
  const int wg      = blockIdx.x;           // 0..31
  const int jt      = wg >> 1;
  const int rowHalf = wg & 1;
  const int rBase   = rowHalf * 128;
  const int gate    = wv >> 3;              // 0..3
  const int rt      = wv & 7;               // 0..7 row tiles
  const int n0      = gate * H_ + jt * 16;  // gemm output column base
  const int ct      = gate * 16 + jt;       // column tile index (0..63)
  const int rr      = rowHalf * 8 + rt;     // row tile within step (0..15)
  const int arow    = rBase + rt * 16 + (lane & 15);

  // register-resident recurrent weights: 8 fragments = 64 VGPRs
  v16bf bfrag[8];
  for (int q = 0; q < 8; ++q)
    bfrag[q] = load_b_frag(Whh, n0, H_, q * 32, lane);

  { // zero c-state
    int idx = threadIdx.x;
    lds_c[idx >> 4][idx & 15] = 0.f;
    idx += 1024;
    lds_c[idx >> 4][idx & 15] = 0.f;
  }
  __syncthreads();

  const float* xpBase =
      xproj + (((size_t)ct * 16 + rr) * 32 + lane) * 8;   // + t*1024*32*8

  for (int t = 0; t < T_; ++t) {
    const unsigned short* hin  = hbuf + (size_t)(t & 1) * (B_ * H_);
    unsigned short*       hout = hbuf + (size_t)((t + 1) & 1) * (B_ * H_);

    // seed accumulator with x_proj[t] tile: one contiguous 32B load
    const float* xp = xpBase + (size_t)t * (1024 * 32 * 8);
    v8f acc = *reinterpret_cast<const v8f*>(xp);
    if (t + 1 < T_)
      __builtin_prefetch(xp + 1024 * 32 * 8, 0, 0);

    // recurrent GEMM: K = 256 -> 8 WMMAs, B operands already in registers
    for (int kk = 0; kk < H_; kk += 32) {
      v16bf a = load_a_frag(hin, arow, H_, kk, lane);
      acc = __builtin_amdgcn_wmma_f32_16x16x32_bf16(
          false, a, false, bfrag[kk >> 5], (short)0, acc, false, false);
    }

    // stage gate tile to LDS
    {
      const int j   = lane & 15;
      const int rl0 = rt * 16 + 8 * (lane >> 4);
      for (int r = 0; r < 8; ++r)
        lds_g[gate][rl0 + r][j] = acc[r];
    }
    __syncthreads();

    // elementwise LSTM cell: 2048 cells / 1024 threads
    for (int u = 0; u < 2; ++u) {
      const int idx = threadIdx.x + u * 1024;
      const int rl = idx >> 4, j = idx & 15;
      const float gi = lds_g[0][rl][j];
      const float gf = lds_g[1][rl][j];
      const float gg = lds_g[2][rl][j];
      const float go = lds_g[3][rl][j];
      const float si = 1.f / (1.f + __expf(-gi));
      const float sf = 1.f / (1.f + __expf(-gf));
      const float tg = 2.f / (1.f + __expf(-2.f * gg)) - 1.f;
      const float so = 1.f / (1.f + __expf(-go));
      const float c  = sf * lds_c[rl][j] + si * tg;
      lds_c[rl][j]   = c;
      const float tc = 2.f / (1.f + __expf(-2.f * c)) - 1.f;
      const float h  = so * tc;
      const int bb   = rBase + rl;
      const int col  = jt * 16 + j;
      hout[(size_t)bb * H_ + col] = f32_bf16(h);
      if (seqOut) seqOut[((size_t)t * B_ + bb) * H_ + col] = f32_bf16(h);
      if (lastH && t == T_ - 1) lastH[(size_t)bb * H_ + col] = h;
    }

    // grid-wide barrier (monotonic counter; memset to 0 before launch)
    __threadfence();
    __syncthreads();
    if (threadIdx.x == 0) {
      atomicAdd(bar, 1u);
      const unsigned int target = 32u * (unsigned)(t + 1);
      while (*(volatile unsigned int*)bar < target)
        __builtin_amdgcn_s_sleep(1);
    }
    __syncthreads();
    __threadfence();
  }
}

// ------------------------------- final FC ----------------------------------
__global__ void final_fc(const float* __restrict__ lastH,
                         const float* __restrict__ Wfc,
                         const float* __restrict__ bfc,
                         float* __restrict__ out) {
  const int b = threadIdx.x;
  float s = 0.f;
  for (int k = 0; k < H_; ++k) s += lastH[(size_t)b * H_ + k] * Wfc[k];
  out[b] = s + bfc[0];
}

// ------------------------------- launcher ----------------------------------
extern "C" void kernel_launch(void* const* d_in, const int* in_sizes, int n_in,
                              void* d_out, int out_size, void* d_ws, size_t ws_size,
                              hipStream_t stream) {
  (void)in_sizes; (void)n_in; (void)out_size; (void)ws_size;

  const float* x    = (const float*)d_in[0];
  const float* Wih0 = (const float*)d_in[1];
  const float* Whh0 = (const float*)d_in[2];
  const float* bih0 = (const float*)d_in[3];
  const float* bhh0 = (const float*)d_in[4];
  const float* Wih1 = (const float*)d_in[5];
  const float* Whh1 = (const float*)d_in[6];
  const float* bih1 = (const float*)d_in[7];
  const float* bhh1 = (const float*)d_in[8];
  const float* Wfc  = (const float*)d_in[9];
  const float* bfc  = (const float*)d_in[10];

  char* ws = (char*)d_ws;
  // workspace layout (bytes)
  unsigned int*   bar   = (unsigned int*)(ws + 0);           //     64
  unsigned short* hbuf  = (unsigned short*)(ws + 64);        // 262144 (2x ping-pong)
  float*          lastH = (float*)(ws + 262208);             // 262144
  unsigned short* x_bf  = (unsigned short*)(ws + 524352);    // 16777216 (T,B,I)
  unsigned short* Wih0b = (unsigned short*)(ws + 17301568);  // 131072
  unsigned short* Whh0b = (unsigned short*)(ws + 17432640);  // 524288
  unsigned short* Wih1b = (unsigned short*)(ws + 17956928);  // 524288
  unsigned short* Whh1b = (unsigned short*)(ws + 18481216);  // 524288
  unsigned short* h1seq = (unsigned short*)(ws + 19005504);  // 67108864 (T,B,H)
  float*          xproj = (float*)(ws + 86114368);           // 536870912 (frag order)

  // zero barrier counter + h ping-pong buffers (graph-capture legal)
  hipMemsetAsync(ws, 0, 64 + 2 * B_ * H_ * 2, stream);

  // bf16 copies of activations/weights (x permuted to t-major rows)
  cvt_permute_x<<<512, 256, 0, stream>>>(x, x_bf);
  cvt_f32_bf16<<<64,  256, 0, stream>>>(Wih0, Wih0b, G_ * I_);
  cvt_f32_bf16<<<256, 256, 0, stream>>>(Whh0, Whh0b, G_ * H_);
  cvt_f32_bf16<<<256, 256, 0, stream>>>(Wih1, Wih1b, G_ * H_);
  cvt_f32_bf16<<<256, 256, 0, stream>>>(Whh1, Whh1b, G_ * H_);

  // layer 0
  xproj_gemm<<<16384, 256, 0, stream>>>(x_bf, Wih0b, bih0, bhh0, xproj, I_);
  lstm_scan<<<32, 1024, 0, stream>>>(xproj, Whh0b, hbuf, h1seq, nullptr, bar);

  // layer 1
  xproj_gemm<<<16384, 256, 0, stream>>>(h1seq, Wih1b, bih1, bhh1, xproj, H_);
  hipMemsetAsync(ws, 0, 64 + 2 * B_ * H_ * 2, stream);
  lstm_scan<<<32, 1024, 0, stream>>>(xproj, Whh1b, hbuf, nullptr, lastH, bar);

  // head
  final_fc<<<1, 256, 0, stream>>>(lastH, Wfc, bfc, (float*)d_out);
}
